// LSTMModel_16681652978098
// MI455X (gfx1250) — matrix operand order repaired and checked by reading, not compiled, not run on hardware
//
#include <hip/hip_runtime.h>
#include <hip/hip_fp16.h>

typedef _Float16 f16;
typedef _Float16 v16h __attribute__((ext_vector_type(16)));
typedef float    v8f  __attribute__((ext_vector_type(8)));

#define B_   512
#define T_   512
#define I_   75
#define IPAD 96      // K padded to 3x32 for WMMA
#define H_   128
#define G4   512     // 4*H
#define NC   10

// LDS row paddings (break 64-bank stride conflicts, keep 16B alignment)
#define XT_PAD 104   // 96  -> 104 halves (208B, 52 dwords)
#define H16P   136   // 128 -> 136 halves (272B, 68 dwords)
#define CP     132   // 128 -> 132 floats

union Frag  { v16h h; uint4 u[2]; };
union Half8 { uint4 u; f16 h[8]; };

__device__ __forceinline__ float sigf(float x)     { return 1.0f / (1.0f + __expf(-x)); }
__device__ __forceinline__ float tanh_fast(float x){ return 2.0f / (1.0f + __expf(-2.0f * x)) - 1.0f; }

// ---------------------------------------------------------------------------
// Convert/pack weights: w_ih -> f16 padded [512][96], w_hh -> f16 [512][128],
// bias = b_ih + b_hh (f32 [512]).
// ---------------------------------------------------------------------------
__global__ void conv_params(const float* __restrict__ w_ih, const float* __restrict__ w_hh,
                            const float* __restrict__ b_ih, const float* __restrict__ b_hh,
                            f16* __restrict__ wih16, f16* __restrict__ whh16,
                            float* __restrict__ bias) {
    int g = blockIdx.x * blockDim.x + threadIdx.x;
    if (g >= G4) return;
    for (int k = 0; k < IPAD; ++k)
        wih16[g * IPAD + k] = (k < I_) ? (f16)w_ih[g * I_ + k] : (f16)0.f;
    for (int k = 0; k < H_; ++k)
        whh16[g * H_ + k] = (f16)w_hh[g * H_ + k];
    bias[g] = b_ih[g] + b_hh[g];
}

// ---------------------------------------------------------------------------
// GEMM1: xgT[t][col][b] = f16( x[b][t] @ w_ih^T + bias ), time-major /
// batch-minor so the recurrent kernel reads 8 consecutive batch rows of one
// gate column as a single uint4. One (t, 16-batch-row) tile per workgroup;
// 8 waves cover the 32 N-tiles (4 each); K = 96 = 3 WMMA frags.
// ---------------------------------------------------------------------------
__global__ __launch_bounds__(256)
void gemm_xw(const float* __restrict__ x, const f16* __restrict__ wih16,
             const float* __restrict__ bias, f16* __restrict__ xgT) {
    __shared__ f16 xt[16][XT_PAD];
    const int tid = threadIdx.x;
    const int t   = blockIdx.x;          // timestep
    const int bb  = blockIdx.y;          // batch block (16 rows)

    // Stage + convert x tile [16 x 75] -> f16, zero-pad K to 96
    {
        const int m  = tid >> 4;
        const int c0 = tid & 15;
        const float* xr = x + ((long)(bb * 16 + m) * T_ + t) * I_;
        for (int c = c0; c < XT_PAD; c += 16)
            xt[m][c] = (c < I_) ? (f16)xr[c] : (f16)0.f;
    }
    __syncthreads();

    const int lane = tid & 31;
    const int wv   = tid >> 5;
    const int n    = lane & 15;
    const int hi   = lane >> 4;

    // A fragments (shared across waves) from LDS: contiguous K = hi*16 + j order
    Frag afr[3];
#pragma unroll
    for (int kt = 0; kt < 3; ++kt) {
        const f16* ab = &xt[n][kt * 32 + hi * 16];
        afr[kt].u[0] = *reinterpret_cast<const uint4*>(ab);        // K = kt*32+hi*16 .. +7
        afr[kt].u[1] = *reinterpret_cast<const uint4*>(ab + 8);    // K = kt*32+hi*16+8 .. +15
    }

#pragma unroll
    for (int nn = 0; nn < 4; ++nn) {
        const int nt  = wv * 4 + nn;
        const int col = nt * 16 + n;
        v8f acc = {0.f, 0.f, 0.f, 0.f, 0.f, 0.f, 0.f, 0.f};
#pragma unroll
        for (int kt = 0; kt < 3; ++kt) {
            Frag bfr;   // B[k][n] = w_ih[col][k]; lane n holds column, K = hi*16 + j
            const f16* bb2 = wih16 + (size_t)col * IPAD + kt * 32 + hi * 16;
            bfr.u[0] = *reinterpret_cast<const uint4*>(bb2);
            bfr.u[1] = *reinterpret_cast<const uint4*>(bb2 + 8);
            acc = __builtin_amdgcn_wmma_f32_16x16x32_f16(false, afr[kt].h, false, bfr.h,
                                                         (short)0, acc, false, false);
        }
        const float bv = bias[col];
        Half8 pk;                                   // D: M = hi*8+r (batch), N = col
#pragma unroll
        for (int r = 0; r < 8; ++r) pk.h[r] = (f16)(acc[r] + bv);
        // xgT[((t*512 + col) * B) + bb*16 + hi*8 .. +7]  (16B aligned)
        *reinterpret_cast<uint4*>(xgT + ((long)t * G4 + col) * B_ + bb * 16 + hi * 8) = pk.u;
    }
}

// ---------------------------------------------------------------------------
// Recurrence: one workgroup per 16 batch rows. Gate-interleaved N-tile
// assignment: wave wv computes tiles {wv, 8+wv, 16+wv, 24+wv} = i/f/g/o for
// hidden block j in [16wv, 16wv+16). Each thread then owns all four gate
// pre-activations of its 8 cells in WMMA accumulators -> cell state c lives
// in registers for all 512 steps, h double-buffers through LDS (f16), and
// only ONE barrier per step is needed. w_hh stays in VGPRs as B-fragments.
// ---------------------------------------------------------------------------
__global__ __launch_bounds__(256)
void lstm_rec(const f16* __restrict__ xgT, const f16* __restrict__ whh16,
              const float* __restrict__ fc_w, const float* __restrict__ fc_b,
              float* __restrict__ out) {
    __shared__ f16   hs[2][16][H16P];  // double-buffered h (WMMA A operand)
    __shared__ float hf[16][CP];       // h in f32 at final step (for FC)

    const int tid  = threadIdx.x;
    const int lane = tid & 31;
    const int wv   = tid >> 5;
    const int n    = lane & 15;
    const int hi   = lane >> 4;
    const long bg  = (long)blockIdx.x * 16;

    for (int i = tid; i < 16 * H16P; i += 256) ((f16*)hs[0])[i] = (f16)0.f;

    // Preload w_hh as WMMA B fragments: rows gate*128 + wv*16 + n, 4 K-frags
    Frag bfr[4][4];
#pragma unroll
    for (int g = 0; g < 4; ++g) {
        const int col = g * H_ + wv * 16 + n;
#pragma unroll
        for (int kt = 0; kt < 4; ++kt) {
            const f16* bb = whh16 + (size_t)col * H_ + kt * 32 + hi * 16;
            bfr[g][kt].u[0] = *reinterpret_cast<const uint4*>(bb);
            bfr[g][kt].u[1] = *reinterpret_cast<const uint4*>(bb + 8);
        }
    }

    float c_reg[8];
#pragma unroll
    for (int r = 0; r < 8; ++r) c_reg[r] = 0.f;

    const int  jcol = wv * 16 + n;           // hidden index owned by this thread
    const int  Mb   = hi * 8;                // first of this thread's 8 batch rows
    // xgT element for (t, gate g, col jcol, batch bg+Mb..+7)
    const f16* xbase = xgT + (long)jcol * B_ + bg + Mb;

    __syncthreads();

    for (int t = 0; t < T_; ++t) {
        const int buf = t & 1;

        // ---- A fragments of h from LDS ----
        Frag afr[4];
#pragma unroll
        for (int kt = 0; kt < 4; ++kt) {
            const f16* ab = &hs[buf][n][kt * 32 + hi * 16];
            afr[kt].u[0] = *reinterpret_cast<const uint4*>(ab);
            afr[kt].u[1] = *reinterpret_cast<const uint4*>(ab + 8);
        }

        // ---- h @ w_hh^T : four gate accumulators per thread ----
        v8f acc[4];
#pragma unroll
        for (int g = 0; g < 4; ++g) {
            v8f a = {0.f, 0.f, 0.f, 0.f, 0.f, 0.f, 0.f, 0.f};
#pragma unroll
            for (int kt = 0; kt < 4; ++kt)
                a = __builtin_amdgcn_wmma_f32_16x16x32_f16(false, afr[kt].h, false,
                                                           bfr[g][kt].h, (short)0, a,
                                                           false, false);
            acc[g] = a;
        }

        // ---- fetch xg for this step: 8 batch rows per gate as one uint4 ----
        const f16* xr = xbase + (long)t * G4 * B_;
        Half8 xh[4];
#pragma unroll
        for (int g = 0; g < 4; ++g)
            xh[g].u = *reinterpret_cast<const uint4*>(xr + (long)g * H_ * B_);

        if (t + 1 < T_) {   // prefetch next timestep's slice
            const f16* nx = xr + (long)G4 * B_;
#pragma unroll
            for (int g = 0; g < 4; ++g)
                __builtin_prefetch(nx + (long)g * H_ * B_, 0, 0);
        }

        // ---- LSTM cell update, fully in registers (torch order i,f,g,o) ----
        const int wbuf = buf ^ 1;
#pragma unroll
        for (int r = 0; r < 8; ++r) {
            const float iv = sigf     (acc[0][r] + (float)xh[0].h[r]);
            const float fv = sigf     (acc[1][r] + (float)xh[1].h[r]);
            const float gv = tanh_fast(acc[2][r] + (float)xh[2].h[r]);
            const float ov = sigf     (acc[3][r] + (float)xh[3].h[r]);
            const float cv = fv * c_reg[r] + iv * gv;
            const float hv = ov * tanh_fast(cv);
            c_reg[r] = cv;
            hs[wbuf][Mb + r][jcol] = (f16)hv;
            if (t == T_ - 1) hf[Mb + r][jcol] = hv;
        }
        __syncthreads();
    }

    // ---- final FC: out[b][k] = h . fc_w[k] + fc_b[k] ----
    if (tid < 16 * NC) {
        const int b = tid / NC, k = tid % NC;
        float s = fc_b[k];
        for (int j = 0; j < H_; ++j) s += hf[b][j] * fc_w[k * H_ + j];
        out[(bg + b) * NC + k] = s;
    }
}

// ---------------------------------------------------------------------------
extern "C" void kernel_launch(void* const* d_in, const int* in_sizes, int n_in,
                              void* d_out, int out_size, void* d_ws, size_t ws_size,
                              hipStream_t stream) {
    const float* x    = (const float*)d_in[0];
    const float* w_ih = (const float*)d_in[1];
    const float* w_hh = (const float*)d_in[2];
    const float* b_ih = (const float*)d_in[3];
    const float* b_hh = (const float*)d_in[4];
    const float* fc_w = (const float*)d_in[5];
    const float* fc_b = (const float*)d_in[6];

    char*  ws  = (char*)d_ws;
    size_t off = 0;
    f16*   xgT   = (f16*)(ws + off); off += (size_t)B_ * T_ * G4 * sizeof(f16); // 268 MB
    f16*   wih16 = (f16*)(ws + off); off += (size_t)G4 * IPAD * sizeof(f16);
    f16*   whh16 = (f16*)(ws + off); off += (size_t)G4 * H_   * sizeof(f16);
    float* bias  = (float*)(ws + off); off += (size_t)G4 * sizeof(float);

    conv_params<<<2, 256, 0, stream>>>(w_ih, w_hh, b_ih, b_hh, wih16, whh16, bias);
    gemm_xw<<<dim3(T_, B_ / 16), 256, 0, stream>>>(x, wih16, bias, xgT);
    lstm_rec<<<B_ / 16, 256, 0, stream>>>(xgT, whh16, fc_w, fc_b, (float*)d_out);
}
